// ETBlock_33895881900640
// MI455X (gfx1250) — matrix operand order repaired and checked by reading, not compiled, not run on hardware
//
#include <hip/hip_runtime.h>
#include <hip/hip_bf16.h>

// ---------------------------------------------------------------------------
// ET block energy for MI455X (gfx1250, wave32).
// All heavy math runs through v_wmma_f32_16x16x32_bf16 (bf16 in, f32 accum).
// logsumexp is fused into the GEMM epilogues (online max/sum on the WMMA
// accumulator registers, butterfly-reduced across the 16 lanes per row).
// Working set (~45MB bf16 staging) is L2-resident (192MB), so WMMA fragments
// are loaded straight from global memory — no LDS staging needed.
// ---------------------------------------------------------------------------

typedef __bf16 bf16_t;
typedef __attribute__((ext_vector_type(16))) __bf16 v16bf;
typedef __attribute__((ext_vector_type(8)))  __bf16 v8bf;
typedef __attribute__((ext_vector_type(8)))  float  v8f;

#define WMMA_BF16(a, b, c) \
  __builtin_amdgcn_wmma_f32_16x16x32_bf16(false, (a), false, (b), (short)0, (c), false, false)

union AFrag { v16bf v; v8bf h[2]; };

__device__ __forceinline__ v8f zero8() {
  v8f z;
#pragma unroll
  for (int i = 0; i < 8; ++i) z[i] = 0.f;
  return z;
}

// A-matrix 16x32 bf16 fragment: lanes 0-15 hold row M=lane with
// K = k0+0..15; lanes 16-31 hold row M=lane-16 with K = k0+16..31.
// Two contiguous 16B loads per lane.
__device__ __forceinline__ v16bf load_a_frag(const bf16_t* __restrict__ src, int ld,
                                             int row0, int k0, int lane) {
  const int r  = row0 + (lane & 15);
  const int kb = k0 + ((lane >> 4) << 4);
  const bf16_t* p = src + (size_t)r * ld + kb;
  AFrag f;
  f.h[0] = *(const v8bf*)(p);       // K = kb + 0..7
  f.h[1] = *(const v8bf*)(p + 8);   // K = kb + 8..15
  return f.v;
}

// B-matrix 32x16 bf16 fragment: lane&15 selects column N; lanes 0-15 carry
// K=k0+0..15, lanes 16-31 carry K=k0+16..31 (contiguous per lane).
// src stored [n][k] row-major, so a column of B is one contiguous 32B load.
__device__ __forceinline__ v16bf load_b_frag(const bf16_t* __restrict__ src, int ld,
                                             int n0, int k0, int lane) {
  const int n  = n0 + (lane & 15);
  const int kb = k0 + ((lane >> 4) << 4);
  return *(const v16bf*)(src + (size_t)n * ld + kb);
}

// Butterfly-combine (max,sum) pairs across the 16 lanes that share each row of
// the C/D tile, then sum lse = m + log(s) over the wave's 16 rows.
// Result valid on lane 0.
__device__ __forceinline__ float lse_reduce(float m[8], float s[8]) {
#pragma unroll
  for (int mask = 1; mask < 16; mask <<= 1) {
#pragma unroll
    for (int v = 0; v < 8; ++v) {
      float mo = __shfl_xor(m[v], mask, 32);
      float so = __shfl_xor(s[v], mask, 32);
      float M  = fmaxf(m[v], mo);
      s[v] = s[v] * __expf(m[v] - M) + so * __expf(mo - M);
      m[v] = M;
    }
  }
  float acc = 0.f;
#pragma unroll
  for (int v = 0; v < 8; ++v) acc += m[v] + __logf(s[v]);
  acc += __shfl_xor(acc, 16, 32);  // rows 0-7 (lane 0) + rows 8-15 (lane 16)
  return acc;
}

// ---------------------------------------------------------------------------
__global__ void cvt_f32_bf16_kernel(const float* __restrict__ in,
                                    bf16_t* __restrict__ out, int n) {
  int i = blockIdx.x * blockDim.x + threadIdx.x;
  const int stride = gridDim.x * blockDim.x;
  for (; i < n; i += stride) out[i] = (bf16_t)in[i];
}

__global__ void zero_out_kernel(float* __restrict__ out) { out[0] = 0.f; }

// ---------------------------------------------------------------------------
// q/k projection: [8192 x 768] = g_bf [8192 x 768] @ w^T, w stored [768][768]
// (row index hz = h*64+z). Output stored [b][h][kk][z] bf16 for the attention
// phase. blockIdx.z selects (wq->q) vs (wk->k). 4 waves/block, one 16x16
// output tile per wave, 24 k-steps of 32 over D=768.
__global__ void proj_qk_kernel(const bf16_t* __restrict__ g,
                               const bf16_t* __restrict__ wq,
                               const bf16_t* __restrict__ wk,
                               bf16_t* __restrict__ qout,
                               bf16_t* __restrict__ kout) {
  const int lane = threadIdx.x & 31;
  const int wave = threadIdx.x >> 5;
  const bf16_t* __restrict__ w = blockIdx.z ? wk : wq;
  bf16_t* __restrict__ out     = blockIdx.z ? kout : qout;

  const int row0 = (blockIdx.x * 4 + wave) * 16;  // in [0, 8192)
  const int col0 = blockIdx.y * 16;               // in [0, 768): h*64+z

  v8f c = zero8();
#pragma unroll 4
  for (int t = 0; t < 24; ++t) {
    v16bf a = load_a_frag(g, 768, row0, t * 32, lane);
    v16bf b = load_b_frag(w, 768, col0, t * 32, lane);
    c = WMMA_BF16(a, b, c);
  }

  // C/D layout: lanes 0-15 VGPR v -> row v, col=lane; lanes 16-31 -> row v+8.
  const int h = col0 >> 6;
  const int z = (col0 & 63) + (lane & 15);
  const int rbase = row0 + ((lane >> 4) << 3);
#pragma unroll
  for (int v = 0; v < 8; ++v) {
    const int r    = rbase + v;
    const int bidx = r >> 10;
    const int kk   = r & 1023;
    out[(((size_t)(bidx * 12 + h) << 10) | kk) * 64 + z] = (bf16_t)c[v];
  }
}

// ---------------------------------------------------------------------------
// Attention energy: per (b,h), S = Q K^T (z=64 inner), online logsumexp over
// the 1024 keys, summed over queries. Q fragments live in registers across
// all 64 key tiles. Adds -1/ATTN_BETA * sum(lse) = -8 * sum(lse) to out.
__global__ void attn_energy_kernel(const bf16_t* __restrict__ qb,
                                   const bf16_t* __restrict__ kb,
                                   float* __restrict__ out) {
  const int lane = threadIdx.x & 31;
  const int wave = threadIdx.x >> 5;
  const size_t bh = (size_t)(blockIdx.z * 12 + blockIdx.y) * 1024 * 64;
  const bf16_t* __restrict__ qh = qb + bh;
  const bf16_t* __restrict__ kh = kb + bh;
  const int qrow0 = blockIdx.x * 64 + wave * 16;

  const v16bf a0 = load_a_frag(qh, 64, qrow0, 0, lane);
  const v16bf a1 = load_a_frag(qh, 64, qrow0, 32, lane);

  float m[8], s[8];
#pragma unroll
  for (int v = 0; v < 8; ++v) { m[v] = -3.0e38f; s[v] = 0.f; }

  for (int kt = 0; kt < 64; ++kt) {
    v16bf b0 = load_b_frag(kh, 64, kt * 16, 0, lane);
    v16bf b1 = load_b_frag(kh, 64, kt * 16, 32, lane);
    v8f c = zero8();
    c = WMMA_BF16(a0, b0, c);
    c = WMMA_BF16(a1, b1, c);
#pragma unroll
    for (int v = 0; v < 8; ++v) {
      float x = 0.125f * c[v];  // ATTN_BETA = 1/sqrt(64)
      float M = fmaxf(m[v], x);
      s[v] = s[v] * __expf(m[v] - M) + __expf(x - M);
      m[v] = M;
    }
  }
  float acc = lse_reduce(m, s);
  if (lane == 0) atomicAdd(out, -8.0f * acc);  // -1/ATTN_BETA
}

// ---------------------------------------------------------------------------
// Hopfield energy: h = beta * g W^T, [8192 x 3072], inner D=768; online
// logsumexp over M per row. 4 N-tiles share each A-fragment load (4x reuse).
// Adds -1/beta * sum(lse) to out.
__global__ void hopfield_energy_kernel(const bf16_t* __restrict__ gb,
                                       const bf16_t* __restrict__ wb,
                                       const float* __restrict__ beta_p,
                                       float* __restrict__ out) {
  const int lane = threadIdx.x & 31;
  const int wave = threadIdx.x >> 5;
  const int row0 = blockIdx.x * 64 + wave * 16;  // in [0, 8192)
  const float beta = beta_p[0];

  float m[8], s[8];
#pragma unroll
  for (int v = 0; v < 8; ++v) { m[v] = -3.0e38f; s[v] = 0.f; }

  for (int nc = 0; nc < 48; ++nc) {  // 192 N-tiles, 4 at a time
    const int n0 = nc * 64;
    v8f cs[4];
#pragma unroll
    for (int j = 0; j < 4; ++j) cs[j] = zero8();

    for (int t = 0; t < 24; ++t) {
      v16bf a = load_a_frag(gb, 768, row0, t * 32, lane);
#pragma unroll
      for (int j = 0; j < 4; ++j) {
        v16bf b = load_b_frag(wb, 768, n0 + j * 16, t * 32, lane);
        cs[j] = WMMA_BF16(a, b, cs[j]);
      }
    }
#pragma unroll
    for (int j = 0; j < 4; ++j) {
#pragma unroll
      for (int v = 0; v < 8; ++v) {
        float x = beta * cs[j][v];
        float M = fmaxf(m[v], x);
        s[v] = s[v] * __expf(m[v] - M) + __expf(x - M);
        m[v] = M;
      }
    }
  }
  float acc = lse_reduce(m, s);
  if (lane == 0) atomicAdd(out, (-1.0f / beta) * acc);
}

// ---------------------------------------------------------------------------
extern "C" void kernel_launch(void* const* d_in, const int* in_sizes, int n_in,
                              void* d_out, int out_size, void* d_ws, size_t ws_size,
                              hipStream_t stream) {
  (void)in_sizes; (void)n_in; (void)out_size; (void)ws_size;
  const float* g    = (const float*)d_in[0];  // [8,1024,768]
  const float* wq   = (const float*)d_in[1];  // [12,64,768]
  const float* wk   = (const float*)d_in[2];  // [12,64,768]
  const float* w_hn = (const float*)d_in[3];  // [3072,768]
  const float* beta = (const float*)d_in[4];  // [1]
  float* out = (float*)d_out;

  const size_t NG = 8ull * 1024 * 768;      // 6,291,456
  const size_t NW = 12ull * 64 * 768;       //   589,824
  const size_t NH = 3072ull * 768;          // 2,359,296
  const size_t NQ = 8ull * 12 * 1024 * 64;  // 6,291,456

  char* base = (char*)d_ws;
  auto align256 = [](size_t x) { return (x + 255) & ~(size_t)255; };
  size_t off = 0;
  bf16_t* g_bf  = (bf16_t*)(base + off); off = align256(off + NG * 2);
  bf16_t* wq_bf = (bf16_t*)(base + off); off = align256(off + NW * 2);
  bf16_t* wk_bf = (bf16_t*)(base + off); off = align256(off + NW * 2);
  bf16_t* wh_bf = (bf16_t*)(base + off); off = align256(off + NH * 2);
  bf16_t* q_bf  = (bf16_t*)(base + off); off = align256(off + NQ * 2);
  bf16_t* k_bf  = (bf16_t*)(base + off); off = align256(off + NQ * 2);

  cvt_f32_bf16_kernel<<<1024, 256, 0, stream>>>(g, g_bf, (int)NG);
  cvt_f32_bf16_kernel<<<256, 256, 0, stream>>>(wq, wq_bf, (int)NW);
  cvt_f32_bf16_kernel<<<256, 256, 0, stream>>>(wk, wk_bf, (int)NW);
  cvt_f32_bf16_kernel<<<1024, 256, 0, stream>>>(w_hn, wh_bf, (int)NH);
  zero_out_kernel<<<1, 1, 0, stream>>>(out);

  // q/k projections: 512 row-tiles / 4 waves, 48 col-tiles, z in {wq, wk}.
  proj_qk_kernel<<<dim3(128, 48, 2), 128, 0, stream>>>(g_bf, wq_bf, wk_bf, q_bf, k_bf);
  // attention energy: 16 q-tiles of 64 rows x 12 heads x 8 batches.
  attn_energy_kernel<<<dim3(16, 12, 8), 128, 0, stream>>>(q_bf, k_bf, out);
  // hopfield energy: 128 row-tiles of 64 rows.
  hopfield_energy_kernel<<<128, 128, 0, stream>>>(g_bf, wh_bf, beta, out);
}
